// LegacySatelliteClassifier_14851996909888
// MI455X (gfx1250) — matrix operand order repaired and checked by reading, not compiled, not run on hardware
//
#include <hip/hip_runtime.h>
#include <stdint.h>

// ---------------------------------------------------------------------------
// Types for CDNA5 WMMA (gfx1250, wave32)
// ---------------------------------------------------------------------------
typedef __bf16 bf16;
typedef bf16  v16bf __attribute__((ext_vector_type(16)));
typedef float v8f   __attribute__((ext_vector_type(8)));

// ---------------------------------------------------------------------------
// Small deterministic RNG (PCG hash + Box-Muller) for the noise / fading parts
// ---------------------------------------------------------------------------
__device__ __forceinline__ uint32_t pcg(uint32_t v) {
    v = v * 747796405u + 2891336453u;
    uint32_t w = ((v >> ((v >> 28u) + 4u)) ^ v) * 277803737u;
    return (w >> 22u) ^ w;
}
__device__ __forceinline__ float u01(uint32_t h) {
    return (float)(h >> 8) * (1.0f / 16777216.0f) + 1e-7f;
}
__device__ __forceinline__ float gaussf(uint32_t s) {
    float u1 = u01(pcg(s));
    float u2 = u01(pcg(s ^ 0x9e3779b9u));
    return __fsqrt_rn(-2.0f * __logf(u1)) * __cosf(6.28318530718f * u2);
}

// ---------------------------------------------------------------------------
// Implicit-GEMM conv kernel: D = A(bf16 weights) x B(bf16 im2col) + epilogue
//   M = Cout (64-tile), N = N*Ho*Wo (64-tile), K = Cin*KH*KW (32-chunks)
//   4 wave32 per block; each wave computes a 16(M) x 64(N) strip with
//   v_wmma_f32_16x16x32_bf16.  Fused BN / bias / residual / relu / sigmoid.
// ---------------------------------------------------------------------------
struct ConvArgs {
    const float* x;
    const bf16*  w;
    float*       out;
    const float* g; const float* b; const float* m; const float* v; // BN (or b = bias)
    const float* residual;
    int N, Ci, Hi, Wi, Co, Ho, Wo, KH, KW, stride, pad;
    int ep;   // 0 = batchnorm, 1 = bias, 2 = none
    int act;  // 0 = none, 1 = relu, 2 = sigmoid
};

__global__ __launch_bounds__(128) void conv_wmma(ConvArgs a) {
    __shared__ bf16 As[64][40];   // [M][K] weight tile (+pad vs bank conflicts)
    __shared__ bf16 Bs[32][72];   // [K][N] im2col tile

    const int tid  = threadIdx.x;
    const int lane = tid & 31;
    const int wv   = tid >> 5;               // wave id 0..3
    const int Ntot = a.N * a.Ho * a.Wo;
    const int Ktot = a.Ci * a.KH * a.KW;
    const int nBase  = blockIdx.x * 64;
    const int coBase = blockIdx.y * 64;
    const int HWo = a.Ho * a.Wo;
    const int KHW = a.KH * a.KW;

    v8f acc[4];
    const v8f vzero = {0.f, 0.f, 0.f, 0.f, 0.f, 0.f, 0.f, 0.f};
#pragma unroll
    for (int s = 0; s < 4; ++s) acc[s] = vzero;

    const bool alignedK = ((Ktot & 7) == 0);   // rows 16B aligned -> async b128 ok

    for (int k0 = 0; k0 < Ktot; k0 += 32) {
        __syncthreads();  // previous iteration's fragments fully consumed

        // ---- A tile: 64 x 32 bf16 weight block, contiguous per row (OIHW) ----
        {
            int row = tid >> 1;
            int cb  = (tid & 1) * 16;
            int co  = coBase + row;
            bool fast = alignedK && (co < a.Co) && (k0 + 32 <= Ktot);
            if (fast) {
                const bf16* src = a.w + (size_t)co * Ktot + k0 + cb;
                uint32_t l0 = (uint32_t)(size_t)(&As[row][cb]);      // low 32b = LDS addr
                uint32_t l1 = (uint32_t)(size_t)(&As[row][cb + 8]);
                uint64_t g0 = (uint64_t)(size_t)src;
                uint64_t g1 = g0 + 16u;
                asm volatile("global_load_async_to_lds_b128 %0, %1, off"
                             :: "v"(l0), "v"(g0) : "memory");
                asm volatile("global_load_async_to_lds_b128 %0, %1, off"
                             :: "v"(l1), "v"(g1) : "memory");
            } else {
#pragma unroll 4
                for (int j = 0; j < 16; ++j) {
                    int kk = k0 + cb + j;
                    bf16 val = (bf16)0.0f;
                    if (co < a.Co && kk < Ktot) val = a.w[(size_t)co * Ktot + kk];
                    As[row][cb + j] = val;
                }
            }
        }

        // ---- B tile: 32(K) x 64(N) gathered im2col with zero padding ----
        {
            int kr = tid >> 2;            // 0..31
            int ns = (tid & 3) * 16;      // 0,16,32,48
            int gk = k0 + kr;
            int ci = 0, fh = 0, fw = 0;
            bool kval = (gk < Ktot);
            if (kval) {
                ci = gk / KHW;
                int r = gk - ci * KHW;
                fh = r / a.KW;
                fw = r - fh * a.KW;
            }
            for (int j = 0; j < 16; ++j) {
                int gn = nBase + ns + j;
                bf16 val = (bf16)0.0f;
                if (kval && gn < Ntot) {
                    int n_ = gn / HWo;
                    int r2 = gn - n_ * HWo;
                    int oh = r2 / a.Wo;
                    int ow = r2 - oh * a.Wo;
                    int ih = oh * a.stride - a.pad + fh;
                    int iw = ow * a.stride - a.pad + fw;
                    if ((unsigned)ih < (unsigned)a.Hi && (unsigned)iw < (unsigned)a.Wi)
                        val = (bf16)a.x[(((size_t)n_ * a.Ci + ci) * a.Hi + ih) * a.Wi + iw];
                }
                Bs[kr][ns + j] = val;
            }
        }

        asm volatile("s_wait_asynccnt 0" ::: "memory");
        __syncthreads();

        // ---- fragments per ISA layouts, then 4 WMMAs (one per 16-col subtile)
        int arow = wv * 16 + (lane & 15);
        int kA   = (lane >> 4) * 8;      // lanes 0-15: K0-7/16-23; 16-31: K8-15/24-31
        v16bf af;
#pragma unroll
        for (int e = 0; e < 16; ++e) af[e] = As[arow][kA + e + (e & 8)];

        int col = lane & 15;
        int kB  = (lane >> 4) * 8;       // lanes 0-15: K0-7/16-23; 16-31: K8-15/24-31
#pragma unroll
        for (int sub = 0; sub < 4; ++sub) {
            v16bf bfr;
#pragma unroll
            for (int e = 0; e < 16; ++e) bfr[e] = Bs[kB + e + (e & 8)][sub * 16 + col];
            acc[sub] = __builtin_amdgcn_wmma_f32_16x16x32_bf16(
                false, af, false, bfr, (short)0, acc[sub], false, false);
        }
    }

    // ---- epilogue: D layout -> (M = r + 8*(lane>=16), N = lane&15) ----
    int colN    = lane & 15;
    int rowBase = (lane >> 4) * 8;
#pragma unroll
    for (int sub = 0; sub < 4; ++sub) {
        int gn = nBase + sub * 16 + colN;
        if (gn >= Ntot) continue;
        int n_ = gn / HWo;
        int r2 = gn - n_ * HWo;
        int oh = r2 / a.Wo;
        int ow = r2 - oh * a.Wo;
#pragma unroll
        for (int r = 0; r < 8; ++r) {
            int co = coBase + wv * 16 + rowBase + r;
            if (co >= a.Co) continue;
            float val = acc[sub][r];
            if (a.ep == 0) {
                val = (val - a.m[co]) * __frsqrt_rn(a.v[co] + 1e-5f) * a.g[co] + a.b[co];
            } else if (a.ep == 1) {
                val += a.b[co];
            }
            size_t oidx = (((size_t)n_ * a.Co + co) * a.Ho + oh) * a.Wo + ow;
            if (a.residual) val += a.residual[oidx];
            if (a.act == 1)      val = fmaxf(val, 0.0f);
            else if (a.act == 2) val = 1.0f / (1.0f + __expf(-val));
            a.out[oidx] = val;
        }
    }
}

// ---------------------------------------------------------------------------
// Helper kernels
// ---------------------------------------------------------------------------
__global__ void cast_bf16_kernel(const float* __restrict__ src, bf16* __restrict__ dst, int n) {
    int i = blockIdx.x * blockDim.x + threadIdx.x;
    if (i < n) dst[i] = (bf16)src[i];
}

// ConvT weights (in,out,3,3) -> conv weights (out,in,3,3) with spatial flip, bf16
__global__ void dec_cast_kernel(const float* __restrict__ w, bf16* __restrict__ dst,
                                int Cin, int Cout) {
    int i = blockIdx.x * blockDim.x + threadIdx.x;
    int total = Cin * Cout * 9;
    if (i >= total) return;
    int o  = i / (Cin * 9);
    int r  = i - o * (Cin * 9);
    int ci = r / 9;
    int rr = r - ci * 9;
    int kh = rr / 3;
    int kw = rr - kh * 3;
    dst[i] = (bf16)w[(((size_t)ci * Cout + o) * 3 + (2 - kh)) * 3 + (2 - kw)];
}

__global__ void maxpool_kernel(const float* __restrict__ in, float* __restrict__ out,
                               int N, int C, int Hi, int Wi, int Ho, int Wo) {
    int idx = blockIdx.x * blockDim.x + threadIdx.x;
    int total = N * C * Ho * Wo;
    if (idx >= total) return;
    int ow = idx % Wo;
    int t  = idx / Wo;
    int oh = t % Ho;  t /= Ho;
    int c  = t % C;
    int n  = t / C;
    float mx = -3.4e38f;
    int ih0 = oh * 2 - 1, iw0 = ow * 2 - 1;
    for (int dh = 0; dh < 3; ++dh) {
        int ih = ih0 + dh;
        if ((unsigned)ih >= (unsigned)Hi) continue;
        for (int dw = 0; dw < 3; ++dw) {
            int iw = iw0 + dw;
            if ((unsigned)iw >= (unsigned)Wi) continue;
            float v = in[(((size_t)n * C + c) * Hi + ih) * Wi + iw];
            mx = fmaxf(mx, v);
        }
    }
    out[idx] = mx;
}

__global__ void gap_kernel(const float* __restrict__ in, float* __restrict__ out,
                           int NC, int HW) {
    int i = blockIdx.x * blockDim.x + threadIdx.x;
    if (i >= NC) return;
    const float* p = in + (size_t)i * HW;
    float s = 0.f;
    for (int j = 0; j < HW; ++j) s += p[j];
    out[i] = s / (float)HW;
}

// out[b][o] = act( dot(in[b,:], w[o,:]) + bias[o] ),  w row-major [Out][In]
__global__ void fc_kernel(const float* __restrict__ in, const float* __restrict__ w,
                          const float* __restrict__ bias, float* __restrict__ out,
                          int B, int In, int Out, int act) {
    int idx = blockIdx.x * blockDim.x + threadIdx.x;
    if (idx >= B * Out) return;
    int o = idx % Out;
    int b = idx / Out;
    const float* xi = in + (size_t)b * In;
    const float* wi = w + (size_t)o * In;
    float s = bias ? bias[o] : 0.f;
    for (int k = 0; k < In; ++k) s += xi[k] * wi[k];
    if (act == 1)      s = fmaxf(s, 0.f);
    else if (act == 2) s = 1.0f / (1.0f + __expf(-s));
    out[idx] = s;
}

// Per-row selection of the pos-th smallest (rank counting) -> mask = (y <= thr)
__global__ __launch_bounds__(512) void thrmask_kernel(const float* __restrict__ y,
                                                      float* __restrict__ mask,
                                                      int C, int pos) {
    __shared__ float sy[512];
    __shared__ float thr;
    int b = blockIdx.x;
    int i = threadIdx.x;
    sy[i] = y[(size_t)b * C + i];
    __syncthreads();
    float vi = sy[i];
    int rank = 0;
    for (int j = 0; j < C; ++j) {
        float vj = sy[j];
        rank += (vj < vi) || (vj == vi && j < i);
    }
    if (rank == pos - 1) thr = vi;
    __syncthreads();
    mask[(size_t)b * C + i] = (vi <= thr) ? 1.0f : 0.0f;
}

__global__ void chscale_kernel(float* __restrict__ h, const float* __restrict__ mask,
                               int C, int HW, size_t total) {
    size_t idx = (size_t)blockIdx.x * blockDim.x + threadIdx.x;
    if (idx >= total) return;
    size_t nc = idx / HW;                // (n*C + c)
    h[idx] *= mask[nc];
}

__global__ void addnoise_kernel(float* __restrict__ h, size_t n, uint32_t seed, float sc) {
    size_t idx = (size_t)blockIdx.x * blockDim.x + threadIdx.x;
    if (idx >= n) return;
    h[idx] += sc * gaussf(seed ^ (uint32_t)idx);
}

// y = (h*x + n)/h = x + n/h (complex, interleaved re/im pairs per row)
__global__ void fading_kernel(float* __restrict__ z, int B, int F, uint32_t seed) {
    int half = F / 2;
    int idx = blockIdx.x * blockDim.x + threadIdx.x;
    if (idx >= B * half) return;
    int b = idx / half;
    int i = idx - b * half;
    uint32_t hb = pcg(seed ^ ((uint32_t)b * 0x9E3779B1u));
    int snr = (int)(hb % 28u);
    float gamma = __powf(10.0f, (float)snr * 0.1f);
    float sigma = __fsqrt_rn(2.0f / gamma);
    uint32_t s0 = seed * 0x85EBCA6Bu ^ (uint32_t)idx;
    float hr = gaussf(s0 ^ 0x1111u);
    float hi = gaussf(s0 ^ 0x2222u);
    float nr = sigma * gaussf(s0 ^ 0x3333u);
    float ni = sigma * gaussf(s0 ^ 0x4444u);
    float d  = hr * hr + hi * hi + 1e-12f;
    float qr = (nr * hr + ni * hi) / d;
    float qi = (ni * hr - nr * hi) / d;
    size_t base = (size_t)b * F + 2 * i;
    z[base]     += qr;
    z[base + 1] += qi;
}

// ---------------------------------------------------------------------------
// Host orchestration
// ---------------------------------------------------------------------------
struct Blk {
    const float* c1; const float* b1[4];
    const float* c2; const float* b2[4];
    const float* dn; const float* dbn[4];
    bool down; int cin, cout, stride;
};

extern "C" void kernel_launch(void* const* d_in, const int* in_sizes, int n_in,
                              void* d_out, int out_size, void* d_ws, size_t ws_size,
                              hipStream_t stream) {
    (void)in_sizes; (void)n_in; (void)out_size;
    int ii = 0;
    auto nextp = [&]() -> const float* { return (const float*)d_in[ii++]; };

    // ---- parse inputs in setup_inputs() insertion order ----
    const float* x = nextp();
    const float* w_conv1 = nextp();
    const float* bn1[4]; for (int i = 0; i < 4; ++i) bn1[i] = nextp();

    Blk blks[8];
    int cfg[4][2] = {{64, 1}, {128, 2}, {256, 2}, {512, 2}};
    int cin = 64, bi = 0;
    for (int s = 0; s < 4; ++s) {
        int cout = cfg[s][0], st = cfg[s][1];
        for (int k = 0; k < 2; ++k) {
            Blk& B = blks[bi++];
            B.cin = (k == 0) ? cin : cout;
            B.cout = cout;
            B.stride = (k == 0) ? st : 1;
            B.down = (k == 0) && (st != 1 || B.cin != cout);
            B.c1 = nextp(); for (int i = 0; i < 4; ++i) B.b1[i] = nextp();
            B.c2 = nextp(); for (int i = 0; i < 4; ++i) B.b2[i] = nextp();
            if (B.down) { B.dn = nextp(); for (int i = 0; i < 4; ++i) B.dbn[i] = nextp(); }
            else B.dn = nullptr;
        }
        cin = cout;
    }
    const float* se_w1 = nextp();
    const float* se_w2 = nextp();
    const float* enc_w[4]; const float* enc_b[4];
    for (int i = 0; i < 4; ++i) { enc_w[i] = nextp(); enc_b[i] = nextp(); }
    const float* dec_w[4]; const float* dec_b[4];
    for (int i = 0; i < 4; ++i) { dec_w[i] = nextp(); dec_b[i] = nextp(); }
    const float* fc_w = nextp();
    const float* fc_b = nextp();

    // ---- carve workspace ----
    const int N = 64;
    char* ws = (char*)d_ws;
    size_t off = 0;
    auto carve = [&](size_t bytes) -> void* {
        void* p = ws + off;
        off += (bytes + 255) & ~(size_t)255;
        return p;
    };
    const size_t bigElems  = (size_t)N * 64 * 112 * 112;   // 51.4M f32
    const size_t stdElems  = (size_t)N * 64 * 56 * 56;     // 12.85M f32
    float* big    = (float*)carve(bigElems * 4);
    float* buf1   = (float*)carve(stdElems * 4);
    float* yv     = (float*)carve((size_t)N * 512 * 4);
    float* y1     = (float*)carve((size_t)N * 32 * 4);
    float* y2     = (float*)carve((size_t)N * 512 * 4);
    float* maskb  = (float*)carve((size_t)N * 512 * 4);
    float* pooled = (float*)carve((size_t)N * 512 * 4);
    const size_t arenaElems = 16u * 1024u * 1024u;
    bf16* arena   = (bf16*)carve(arenaElems * 2);
    if (off > ws_size) return;                // workspace too small: bail safely
    float* buf2 = big;                        // big is dead after maxpool; reuse
    float* buf3 = big + stdElems;
    float* bufI = big + 2 * stdElems;

    // ---- bf16 weight conversion into arena ----
    size_t aoff = 0;
    auto wcast = [&](const float* src, int n) -> const bf16* {
        bf16* dst = arena + aoff;
        aoff += (size_t)((n + 127) & ~127);
        cast_bf16_kernel<<<(n + 255) / 256, 256, 0, stream>>>(src, dst, n);
        return dst;
    };
    auto wdec = [&](const float* src, int ci, int co) -> const bf16* {
        int n = ci * co * 9;
        bf16* dst = arena + aoff;
        aoff += (size_t)((n + 127) & ~127);
        dec_cast_kernel<<<(n + 255) / 256, 256, 0, stream>>>(src, dst, ci, co);
        return dst;
    };

    // ---- conv launcher (square spatial throughout) ----
    auto conv = [&](const float* in, const bf16* wq, float* out,
                    int Ci, int Hi, int Co, int KH, int st, int pad,
                    const float* const* bn, const float* bias,
                    const float* residual, int act) -> int {
        ConvArgs A{};
        A.x = in; A.w = wq; A.out = out; A.residual = residual;
        A.N = N; A.Ci = Ci; A.Hi = Hi; A.Wi = Hi; A.Co = Co;
        A.KH = KH; A.KW = KH; A.stride = st; A.pad = pad;
        A.Ho = (Hi + 2 * pad - KH) / st + 1;
        A.Wo = A.Ho;
        if (bn) { A.ep = 0; A.g = bn[0]; A.b = bn[1]; A.m = bn[2]; A.v = bn[3]; }
        else if (bias) { A.ep = 1; A.b = bias; }
        else A.ep = 2;
        A.act = act;
        int Ntot = N * A.Ho * A.Wo;
        dim3 grid((Ntot + 63) / 64, (Co + 63) / 64);
        conv_wmma<<<grid, 128, 0, stream>>>(A);
        return A.Ho;
    };

    // ================= ResNet-18 trunk =================
    const bf16* q_conv1 = wcast(w_conv1, 64 * 3 * 49);
    conv(x, q_conv1, big, 3, 224, 64, 7, 2, 3, bn1, nullptr, nullptr, 1); // -> 112x112
    {
        int total = N * 64 * 56 * 56;
        maxpool_kernel<<<(total + 255) / 256, 256, 0, stream>>>(big, buf1, N, 64, 112, 112, 56, 56);
    }

    float* cur = buf1; float* t1 = buf2; float* t2 = buf3;
    int H = 56;
    for (int b = 0; b < 8; ++b) {
        Blk& B = blks[b];
        const bf16* q1 = wcast(B.c1, B.cout * B.cin * 9);
        const bf16* q2 = wcast(B.c2, B.cout * B.cout * 9);
        int Ho = conv(cur, q1, t1, B.cin, H, B.cout, 3, B.stride, 1, B.b1, nullptr, nullptr, 1);
        const float* idt = cur;
        if (B.down) {
            const bf16* qd = wcast(B.dn, B.cout * B.cin);
            conv(cur, qd, bufI, B.cin, H, B.cout, 1, B.stride, 0, B.dbn, nullptr, nullptr, 0);
            idt = bufI;
        }
        conv(t1, q2, t2, B.cout, Ho, B.cout, 3, 1, 1, B.b2, nullptr, idt, 1);
        float* old = cur; cur = t2; t2 = old;
        H = Ho;
    }
    // cur: [64, 512, 7, 7]

    // ================= SE block + threshold mask =================
    gap_kernel<<<(N * 512 + 255) / 256, 256, 0, stream>>>(cur, yv, N * 512, 49);
    fc_kernel<<<(N * 32 + 255) / 256, 256, 0, stream>>>(yv, se_w1, nullptr, y1, N, 512, 32, 1);
    fc_kernel<<<(N * 512 + 255) / 256, 256, 0, stream>>>(y1, se_w2, nullptr, y2, N, 32, 512, 2);
    thrmask_kernel<<<N, 512, 0, stream>>>(y2, maskb, 512, 410);
    {
        size_t total = (size_t)N * 512 * 49;
        chscale_kernel<<<(unsigned)((total + 255) / 256), 256, 0, stream>>>(cur, maskb, 512, 49, total);
    }

    // ================= Encoder =================
    int encC[4] = {256, 128, 64, 32};
    int ch = 512;
    for (int i = 0; i < 4; ++i) {
        const bf16* q = wcast(enc_w[i], encC[i] * ch * 9);
        conv(cur, q, t1, ch, 7, encC[i], 3, 1, 1, nullptr, enc_b[i], nullptr, 1);
        float* tmp = cur; cur = t1; t1 = tmp;
        ch = encC[i];
    }

    // ================= Noise + fading channel =================
    {
        size_t nelem = (size_t)N * 32 * 49;
        addnoise_kernel<<<(unsigned)((nelem + 255) / 256), 256, 0, stream>>>(cur, nelem, 0x1337BEEFu, 0.1f);
        int F = 32 * 49, half = F / 2;
        fading_kernel<<<(N * half + 255) / 256, 256, 0, stream>>>(cur, N, F, 0xC0FFEEu);
    }

    // ================= Decoder (ConvT k3 s1 p1 == conv with flipped/transposed W)
    int decIn[4]  = {32, 64, 128, 256};
    int decOut[4] = {64, 128, 256, 512};
    for (int i = 0; i < 4; ++i) {
        const bf16* q = wdec(dec_w[i], decIn[i], decOut[i]);
        conv(cur, q, t1, decIn[i], 7, decOut[i], 3, 1, 1, nullptr, dec_b[i], nullptr,
             (i == 3) ? 2 : 1);
        float* tmp = cur; cur = t1; t1 = tmp;
    }

    // ================= Head =================
    gap_kernel<<<(N * 512 + 255) / 256, 256, 0, stream>>>(cur, pooled, N * 512, 49);
    fc_kernel<<<(N * 45 + 255) / 256, 256, 0, stream>>>(pooled, fc_w, fc_b,
                                                        (float*)d_out, N, 512, 45, 0);
}
